// RA_RA_17686675325051
// MI455X (gfx1250) — matrix operand order repaired and checked by reading, not compiled, not run on hardware
//
#include <hip/hip_runtime.h>
#include <hip/hip_bf16.h>

// B=4, H=W=C=64, NC=19.  N = 4096 rows per batch, D = 64 dims.

typedef __attribute__((ext_vector_type(16))) __bf16 v16bf;
typedef __attribute__((ext_vector_type(8)))  float  v8f;

#define WMMA_BF16(A, B, C) \
  __builtin_amdgcn_wmma_f32_16x16x32_bf16(false, (A), false, (B), (short)0, (C), false, false)

#define USE_ASYNC_LDS 1

// Fast sigmoid(scale*x) with the scale pre-folded into the exp2 constant:
//   sigmoid(s*x) = rcp(1 + exp2(x * (-s*log2(e))))
// One v_mul + v_exp + v_add + v_rcp per element; plenty of precision for
// scores that feed a bf16 WMMA.
__device__ __forceinline__ float fast_sigmoid_pre(float x, float neg_scale_log2e) {
  return __builtin_amdgcn_rcpf(1.0f + __builtin_amdgcn_exp2f(x * neg_scale_log2e));
}

#define LOG2E 1.44269504088896340736f

// ---------------------------------------------------------------------------
// Projection: q = x @ Wq + bq ; v = x @ Wv + bv   (per-batch 4096x64 @ 64x64)
// mode 0 (height pass): xs[k][c] = X[b][k][slab][c]   (row n = slab*64+c, k=h)
// mode 1 (width  pass): xs[k][c] = X[b][slab][k][c]   (row n = slab*64+c, k=w)
// Outputs bf16 [b][n][d].
// ---------------------------------------------------------------------------
__global__ void proj_kernel(const float* __restrict__ X,
                            const float* __restrict__ Wq, const float* __restrict__ bq,
                            const float* __restrict__ Wv, const float* __restrict__ bv,
                            __bf16* __restrict__ Qout, __bf16* __restrict__ Vout,
                            int mode) {
  __shared__ float xs[64][65];
  __shared__ float wqs[64][64];
  __shared__ float wvs[64][64];
  __shared__ float bqs[64];
  __shared__ float bvs[64];

  const int t    = threadIdx.x;          // 0..255
  const int blk  = blockIdx.x;           // b*64 + slab
  const int b    = blk >> 6;
  const int slab = blk & 63;

  for (int i = t; i < 4096; i += 256) {
    wqs[i >> 6][i & 63] = Wq[i];
    wvs[i >> 6][i & 63] = Wv[i];
  }
  if (t < 64) { bqs[t] = bq[t]; bvs[t] = bv[t]; }
  for (int i = t; i < 4096; i += 256) {
    const int k = i >> 6, c = i & 63;
    int idx;
    if (mode == 0) idx = (((b * 64 + k) * 64 + slab) * 64) + c;   // X[b][k][slab][c]
    else           idx = (((b * 64 + slab) * 64 + k) * 64) + c;   // X[b][slab][k][c]
    xs[k][c] = X[idx];
  }
  __syncthreads();

  const int c  = t >> 2;
  const int d0 = (t & 3) * 16;
  float aq[16], av[16];
#pragma unroll
  for (int j = 0; j < 16; ++j) { aq[j] = bqs[d0 + j]; av[j] = bvs[d0 + j]; }
  for (int k = 0; k < 64; ++k) {
    const float xv = xs[k][c];
#pragma unroll
    for (int j = 0; j < 16; ++j) {
      aq[j] += xv * wqs[k][d0 + j];
      av[j] += xv * wvs[k][d0 + j];
    }
  }
  const int base = ((b * 4096 + slab * 64 + c) * 64) + d0;
#pragma unroll
  for (int j = 0; j < 16; ++j) {
    Qout[base + j] = (__bf16)aq[j];
    Vout[base + j] = (__bf16)av[j];
  }
}

// ---------------------------------------------------------------------------
// pconv[b][h][w] = sum_nc (1 - sigmoid(predict[b][h][w][nc])) * conv_w[nc] + conv_b
// ---------------------------------------------------------------------------
__global__ void pconv_kernel(const float* __restrict__ predict,
                             const float* __restrict__ cw, const float* __restrict__ cb,
                             float* __restrict__ pc) {
  const int i = blockIdx.x * 256 + threadIdx.x;    // 0 .. 4*64*64-1
  if (i >= 4 * 64 * 64) return;
  const float* p = predict + i * 19;
  float acc = cb[0];
#pragma unroll
  for (int j = 0; j < 19; ++j) {
    acc += (1.0f - fast_sigmoid_pre(p[j], -LOG2E)) * cw[j];
  }
  pc[i] = acc;
}

// ---------------------------------------------------------------------------
// Streaming sigmoid-attention:  O[n][d] = sum_m sigmoid(Q[n].Q[m] / 8) * V[m][d]
// One workgroup = 64-row x 64-dim output tile for one (b, slab).  8 waves:
// wave w -> row block r = w>>1 (16 rows), dim half ch = w&1 (32 dims).
// Inner loop streams 32-key tiles of K(=Q) and V through LDS via async copies.
// mode 0: out = x1 = gate*O + add0,               addressed [b][d][slab][c]
// mode 1: out = (gate*O + add0 + add1)*pconv,     addressed [b][slab][d][c]
// ---------------------------------------------------------------------------
__global__ void attn_kernel(const __bf16* __restrict__ Q, const __bf16* __restrict__ V,
                            const float* __restrict__ gate,
                            const float* __restrict__ add0, const float* __restrict__ add1,
                            const float* __restrict__ pconv, float* __restrict__ out,
                            int mode) {
  __shared__ __align__(16) __bf16 Qs[64][64];        // 8 KB: this tile's query rows
  __shared__ __align__(16) __bf16 Ks[32][64];        // 4 KB: streamed key tile
  __shared__ __align__(16) __bf16 Vs[32][64];        // 4 KB: streamed value tile
  __shared__ __align__(16) __bf16 Ss[8][16][32];     // 8 KB: per-wave score relayout

  const int t    = threadIdx.x;
  const int lane = t & 31;
  const int wv   = t >> 5;
  const int b    = blockIdx.x >> 6;
  const int slab = blockIdx.x & 63;

  const __bf16* Qb = Q + (size_t)b * 4096 * 64;
  const __bf16* Vb = V + (size_t)b * 4096 * 64;

  // Stage this tile's 64 query rows (8 KB = 512 x uint4).
  {
    const uint4* src = (const uint4*)(Qb + (size_t)slab * 64 * 64);
    uint4* dst = (uint4*)(&Qs[0][0]);
    dst[t]       = src[t];
    dst[t + 256] = src[t + 256];
  }
  __syncthreads();

  const int r    = wv >> 1;          // row block 0..3
  const int ch   = wv & 1;           // dim half 0..1
  const int m_lo = lane & 15;
  const int hi   = lane >> 4;        // 0: lanes 0-15, 1: lanes 16-31

  // A-fragments for this wave's 16 query rows (K = 0..31 and 32..63).
  // A k order used here (contiguous, matched by B): lane<16 holds K 0..15 ; lane>=16 holds K 16..31.
  v16bf aq0, aq1;
  {
    const int m  = 16 * r + m_lo;
    const int kb = hi ? 16 : 0;
#pragma unroll
    for (int i = 0; i < 16; ++i) {
      const int k = kb + i;
      aq0[i] = Qs[m][k];
      aq1[i] = Qs[m][32 + k];
    }
  }

  // LDS byte offsets for async global->LDS staging (flat LDS addr low 32 bits
  // are the LDS offset per ISA 10.2).
  const unsigned kofs = (unsigned)(uintptr_t)(&Ks[0][0]) + (unsigned)t * 16u;
  const unsigned vofs = (unsigned)(uintptr_t)(&Vs[0][0]) + (unsigned)t * 16u;

  v8f o0 = {0.f, 0.f, 0.f, 0.f, 0.f, 0.f, 0.f, 0.f};
  v8f o1 = {0.f, 0.f, 0.f, 0.f, 0.f, 0.f, 0.f, 0.f};
  const float gg = gate[0];

  for (int kt = 0; kt < 128; ++kt) {
    __syncthreads();   // previous tile fully consumed before overwriting Ks/Vs
    // Stage 32 key rows + 32 value rows (4 KB each = 256 x 16B per array).
#if USE_ASYNC_LDS
    {
      const unsigned long long ka =
          (unsigned long long)(uintptr_t)(Qb + (size_t)kt * 2048) + (unsigned long long)t * 16u;
      const unsigned long long va =
          (unsigned long long)(uintptr_t)(Vb + (size_t)kt * 2048) + (unsigned long long)t * 16u;
      asm volatile("global_load_async_to_lds_b128 %0, %1, off\n\t"
                   "global_load_async_to_lds_b128 %2, %3, off"
                   :
                   : "v"(kofs), "v"(ka), "v"(vofs), "v"(va)
                   : "memory");
      // Pull the next tile toward the WGP caches while this one computes
      // (locality 3 -> WGP-scope prefetch, all cache levels).
      if (kt + 1 < 128) {
        __builtin_prefetch((const char*)(Qb + (size_t)(kt + 1) * 2048) + t * 16, 0, 3);
        __builtin_prefetch((const char*)(Vb + (size_t)(kt + 1) * 2048) + t * 16, 0, 3);
      }
      asm volatile("s_wait_asynccnt 0x0" ::: "memory");
    }
#else
    {
      const uint4* ksrc = (const uint4*)(Qb + (size_t)kt * 2048);
      const uint4* vsrc = (const uint4*)(Vb + (size_t)kt * 2048);
      ((uint4*)&Ks[0][0])[t] = ksrc[t];
      ((uint4*)&Vs[0][0])[t] = vsrc[t];
    }
#endif
    __syncthreads();

    // ---- scores: S(16 rows x 32 keys) = Qrows(16x64) . Ktile(32x64)^T -----
    // B k order used here (contiguous, same as A): lane<16 holds K 0..15, lane>=16 holds K 16..31; col n = key.
    const int kbB = hi ? 16 : 0;
    v8f s0 = {0.f, 0.f, 0.f, 0.f, 0.f, 0.f, 0.f, 0.f};
    v8f s1 = {0.f, 0.f, 0.f, 0.f, 0.f, 0.f, 0.f, 0.f};
    {
      v16bf b0, b1;
      int key = m_lo;                           // key group 0 (keys 0..15)
#pragma unroll
      for (int i = 0; i < 16; ++i) { b0[i] = Ks[key][kbB + i]; b1[i] = Ks[key][32 + kbB + i]; }
      s0 = WMMA_BF16(aq0, b0, s0);
      s0 = WMMA_BF16(aq1, b1, s0);
      key = 16 + m_lo;                          // key group 1 (keys 16..31)
#pragma unroll
      for (int i = 0; i < 16; ++i) { b0[i] = Ks[key][kbB + i]; b1[i] = Ks[key][32 + kbB + i]; }
      s1 = WMMA_BF16(aq0, b0, s1);
      s1 = WMMA_BF16(aq1, b1, s1);
    }

    // sigmoid(s / sqrt(64)); 0.125 scale folded into the exp2 constant.
#pragma unroll
    for (int j = 0; j < 8; ++j) {
      s0[j] = fast_sigmoid_pre(s0[j], -0.125f * LOG2E);
      s1[j] = fast_sigmoid_pre(s1[j], -0.125f * LOG2E);
    }

    // Relayout scores D-frag -> A-frag via per-wave LDS (bf16).
    // D layout: lane col n = m_lo (+16 for s1), row m = j + 8*hi.
    {
#pragma unroll
      for (int j = 0; j < 8; ++j) {
        const int m = j + 8 * hi;
        Ss[wv][m][m_lo]      = (__bf16)s0[j];
        Ss[wv][m][16 + m_lo] = (__bf16)s1[j];
      }
    }
    asm volatile("s_wait_dscnt 0" ::: "memory");   // same-wave LDS RAW fence

    v16bf as;
    {
      const int kb = hi ? 16 : 0;
#pragma unroll
      for (int i = 0; i < 16; ++i) {
        const int k = kb + i;
        as[i] = Ss[wv][m_lo][k];
      }
    }

    // ---- output: O(16x32 dims) += S(16x32 keys) . Vtile(32keys x dims) ----
    {
      const int dim0 = ch * 32 + m_lo;
      const int kbV  = hi ? 16 : 0;
      v16bf bv0, bv1;
#pragma unroll
      for (int i = 0; i < 16; ++i) {
        bv0[i] = Vs[kbV + i][dim0];
        bv1[i] = Vs[kbV + i][dim0 + 16];
      }
      o0 = WMMA_BF16(as, bv0, o0);
      o1 = WMMA_BF16(as, bv1, o1);
    }
  }

  // Epilogue: fuse gate + residual(s) (+ pconv gating for mode 1).
#pragma unroll
  for (int sub = 0; sub < 2; ++sub) {
    const v8f oo = sub ? o1 : o0;
    const int d  = ch * 32 + sub * 16 + m_lo;      // output feature dim (h or w)
#pragma unroll
    for (int j = 0; j < 8; ++j) {
      const int c = 16 * r + j + 8 * hi;           // row within 64-row tile
      const float val = gg * oo[j];
      if (mode == 0) {
        const int idx = (((b * 64 + d) * 64 + slab) * 64) + c;   // x1[b][h=d][w=slab][c]
        out[idx] = val + add0[idx];
      } else {
        const int idx = (((b * 64 + slab) * 64 + d) * 64) + c;   // out[b][h=slab][w=d][c]
        out[idx] = (val + add0[idx] + add1[idx]) * pconv[(b * 64 + slab) * 64 + d];
      }
    }
  }
}

// ---------------------------------------------------------------------------
extern "C" void kernel_launch(void* const* d_in, const int* in_sizes, int n_in,
                              void* d_out, int out_size, void* d_ws, size_t ws_size,
                              hipStream_t stream) {
  const float* feature = (const float*)d_in[0];
  const float* predict = (const float*)d_in[1];
  const float* hq_w    = (const float*)d_in[2];
  const float* hq_b    = (const float*)d_in[3];
  const float* hv_w    = (const float*)d_in[4];
  const float* hv_b    = (const float*)d_in[5];
  const float* wq_w    = (const float*)d_in[6];
  const float* wq_b    = (const float*)d_in[7];
  const float* wv_w    = (const float*)d_in[8];
  const float* wv_b    = (const float*)d_in[9];
  const float* h_gate  = (const float*)d_in[10];
  const float* w_gate  = (const float*)d_in[11];
  const float* conv_w  = (const float*)d_in[12];
  const float* conv_b  = (const float*)d_in[13];
  float* out = (float*)d_out;

  char* ws = (char*)d_ws;
  float*  x1 = (float*)ws;                               // 4 MB  [B,H,W,C]
  __bf16* Qb = (__bf16*)(ws + (4u << 20));               // 2 MB  bf16 [B,4096,64]
  __bf16* Vb = (__bf16*)(ws + (6u << 20));               // 2 MB
  float*  pc = (float*)(ws + (8u << 20));                // 64 KB [B,H,W]

  // Height pass: rows n = w*64+c, feature dim = h.
  proj_kernel <<<256, 256, 0, stream>>>(feature, hq_w, hq_b, hv_w, hv_b, Qb, Vb, 0);
  pconv_kernel<<< 64, 256, 0, stream>>>(predict, conv_w, conv_b, pc);
  attn_kernel <<<256, 256, 0, stream>>>(Qb, Vb, h_gate, feature, nullptr, nullptr, x1, 0);

  // Width pass: rows n = h*64+c, feature dim = w (workspace Q/V reused serially).
  proj_kernel <<<256, 256, 0, stream>>>(x1, wq_w, wq_b, wv_w, wv_b, Qb, Vb, 1);
  attn_kernel <<<256, 256, 0, stream>>>(Qb, Vb, w_gate, feature, x1, pc, out, 1);
}
